// TaskAlignedAssigner_22522808500279
// MI455X (gfx1250) — matrix operand order repaired and checked by reading, not compiled, not run on hardware
//
#include <hip/hip_runtime.h>
#include <hip/hip_bf16.h>

typedef __attribute__((ext_vector_type(16))) _Float16 v16h;
typedef __attribute__((ext_vector_type(8)))  float    v8f;

#define B_DIM   16
#define A_DIM   8400
#define C_DIM   80
#define G_DIM   64
#define K_TOP   13
#define NTILES  (A_DIM / 16)      // 525 full 16-anchor tiles
#define NSPLIT  4                 // anchor splits (blocks) per (b, g-group)
#define NWAVE   8
#define CAND_PER_G (NWAVE * 2 * K_TOP)   // 208 candidates per g per block
#define NCAND_FINAL (NSPLIT * K_TOP)     // 52 candidates per g in final merge

__device__ __forceinline__ float neg_inf() { return -__builtin_inff(); }

__device__ __forceinline__ void argmax_upd(float& bv, int& bi, int& bo,
                                           float ov, int oi, int oo) {
  if (ov > bv || (ov == bv && oi < bi)) { bv = ov; bi = oi; bo = oo; }
}

template <int BASE>
__device__ __forceinline__ void pack8(v16h& av, const float4& f0, const float4& f1) {
  av[BASE + 0] = (_Float16)f0.x; av[BASE + 1] = (_Float16)f0.y;
  av[BASE + 2] = (_Float16)f0.z; av[BASE + 3] = (_Float16)f0.w;
  av[BASE + 4] = (_Float16)f1.x; av[BASE + 5] = (_Float16)f1.y;
  av[BASE + 6] = (_Float16)f1.z; av[BASE + 7] = (_Float16)f1.w;
}

// ---------------------------------------------------------------------------
// Kernel A: metric = score_gather(WMMA one-hot) * IoU^6 * mask, then top-13
// per (b, g, anchor-split).  Grid: 256 blocks (16 b x 4 g-groups x 4 splits),
// 256 threads (8 waves).  Wave handles 16-anchor x 16-gt tiles.
// ---------------------------------------------------------------------------
__global__ __launch_bounds__(256) void tal_metric_topk(
    const float* __restrict__ pred_scores,   // [B,A,C]
    const float* __restrict__ pred_bboxes,   // [B,A,4]
    const int*   __restrict__ gt_labels,     // [B,G]
    const float* __restrict__ gt_bboxes,     // [B,G,4]
    const float* __restrict__ mask_gt,       // [B,G]
    float* __restrict__ topv,                // [B,G,NSPLIT,K_TOP]
    int*   __restrict__ topi)                // [B,G,NSPLIT,K_TOP]
{
  __shared__ float s_cv[16][CAND_PER_G];
  __shared__ int   s_ci[16][CAND_PER_G];
  __shared__ float s_gt[16][4];
  __shared__ int   s_lab[16];
  __shared__ float s_msk[16];

  const int blk   = blockIdx.x;
  const int split = blk & 3;
  const int gg    = (blk >> 2) & 3;
  const int b     = blk >> 4;
  const int gbase = gg * 16;

  const int tid  = threadIdx.x;
  const int w    = tid >> 5;
  const int lane = tid & 31;
  const int gl   = lane & 15;   // local gt index owned by this lane (D-layout N)
  const int hi   = lane >> 4;
  const int hi8  = hi * 8;

  if (tid < 16) {
    const int gi = b * G_DIM + gbase + tid;
    s_gt[tid][0] = gt_bboxes[gi * 4 + 0];
    s_gt[tid][1] = gt_bboxes[gi * 4 + 1];
    s_gt[tid][2] = gt_bboxes[gi * 4 + 2];
    s_gt[tid][3] = gt_bboxes[gi * 4 + 3];
    s_lab[tid]   = gt_labels[gi];
    s_msk[tid]   = mask_gt[gi];
  }
  __syncthreads();

  const float gx0 = s_gt[gl][0], gy0 = s_gt[gl][1];
  const float gx1 = s_gt[gl][2], gy1 = s_gt[gl][3];
  const float garea = (gx1 - gx0) * (gy1 - gy0);
  const int   lab = s_lab[gl];
  const float msk = s_msk[gl];

  // One-hot B operands (32x16 f16) for the 3 K-chunks of 32 classes.
  // B layout: lane gives N=g; element e of v16h gives K = hi*8 + e (e<8), 16 + hi*8 + (e-8) (e>=8).
  v16h bv0, bv1, bv2;
#pragma unroll
  for (int e = 0; e < 16; ++e) {
    const int k = (e < 8) ? (hi * 8 + e) : (16 + hi * 8 + (e - 8));
    bv0[e] = (lab == k)        ? (_Float16)1.0f : (_Float16)0.0f;
    bv1[e] = (lab == (32 + k)) ? (_Float16)1.0f : (_Float16)0.0f;
    bv2[e] = (lab == (64 + k)) ? (_Float16)1.0f : (_Float16)0.0f;
  }

  // Per-lane top-13 (sorted desc; ties keep earlier = smaller anchor idx).
  float tv[K_TOP];
  int   ti[K_TOP];
#pragma unroll
  for (int k = 0; k < K_TOP; ++k) { tv[k] = neg_inf(); ti[k] = 0x7fffffff; }

  for (int T = split + 4 * w; T < NTILES; T += NSPLIT * NWAVE) {
    const int a0 = T * 16;

    // Each lane loads box of anchor a0 + (lane%16); shared via shuffles.
    const float4 myBox =
        *(const float4*)(pred_bboxes + ((size_t)b * A_DIM + a0 + gl) * 4);

    const float* rowp = pred_scores + ((size_t)b * A_DIM + a0 + gl) * C_DIM;
    if (T + NSPLIT * NWAVE < NTILES)
      __builtin_prefetch(rowp + (size_t)(NSPLIT * NWAVE * 16) * C_DIM, 0, 3);

    v8f acc = {};
    // ---- chunk 0: classes 0..31 ----
    {
      v16h av;
      const float4 f0 = *(const float4*)(rowp + hi8);
      const float4 f1 = *(const float4*)(rowp + hi8 + 4);
      const float4 f2 = *(const float4*)(rowp + 16 + hi8);
      const float4 f3 = *(const float4*)(rowp + 16 + hi8 + 4);
      pack8<0>(av, f0, f1);
      pack8<8>(av, f2, f3);
      acc = __builtin_amdgcn_wmma_f32_16x16x32_f16(false, av, false, bv0,
                                                   (short)0, acc, false, false);
    }
    // ---- chunk 1: classes 32..63 ----
    {
      v16h av;
      const float4 f0 = *(const float4*)(rowp + 32 + hi8);
      const float4 f1 = *(const float4*)(rowp + 32 + hi8 + 4);
      const float4 f2 = *(const float4*)(rowp + 48 + hi8);
      const float4 f3 = *(const float4*)(rowp + 48 + hi8 + 4);
      pack8<0>(av, f0, f1);
      pack8<8>(av, f2, f3);
      acc = __builtin_amdgcn_wmma_f32_16x16x32_f16(false, av, false, bv1,
                                                   (short)0, acc, false, false);
    }
    // ---- chunk 2: classes 64..79 (80..95 are zero pad) ----
    {
      v16h av;
      const float4 f0 = *(const float4*)(rowp + 64 + hi8);
      const float4 f1 = *(const float4*)(rowp + 64 + hi8 + 4);
      pack8<0>(av, f0, f1);
#pragma unroll
      for (int e = 8; e < 16; ++e) av[e] = (_Float16)0.0f;
      acc = __builtin_amdgcn_wmma_f32_16x16x32_f16(false, av, false, bv2,
                                                   (short)0, acc, false, false);
    }

    // D layout: VGPR j, this lane -> anchor M = hi*8 + j, gt N = gl.
#pragma unroll
    for (int j = 0; j < 8; ++j) {
      const int src = hi8 + j;  // lane (0..15) holding this anchor's box
      const float bx0 = __shfl(myBox.x, src, 32);
      const float by0 = __shfl(myBox.y, src, 32);
      const float bx1 = __shfl(myBox.z, src, 32);
      const float by1 = __shfl(myBox.w, src, 32);

      const float lx = fmaxf(gx0, bx0), ly = fmaxf(gy0, by0);
      const float rx = fminf(gx1, bx1), ry = fminf(gy1, by1);
      const float ww = fmaxf(rx - lx, 0.0f), hh = fmaxf(ry - ly, 0.0f);
      const float ov = ww * hh;
      const float a2 = (bx1 - bx0) * (by1 - by0);
      const float iou = ov / (garea + a2 - ov + 1e-6f);
      const float i2 = iou * iou;
      const float met = acc[j] * (i2 * i2 * i2) * msk;
      const int aidx = a0 + hi8 + j;

      if (met > tv[K_TOP - 1]) {
        tv[K_TOP - 1] = met;
        ti[K_TOP - 1] = aidx;
#pragma unroll
        for (int q = K_TOP - 1; q > 0; --q) {
          if (tv[q] > tv[q - 1]) {
            const float fv = tv[q]; tv[q] = tv[q - 1]; tv[q - 1] = fv;
            const int fi = ti[q]; ti[q] = ti[q - 1]; ti[q - 1] = fi;
          }
        }
      }
    }
  }

  // Dump per-lane candidate lists into LDS, keyed by local gt.
  {
    const int sb = w * (2 * K_TOP) + hi * K_TOP;
#pragma unroll
    for (int k = 0; k < K_TOP; ++k) {
      s_cv[gl][sb + k] = tv[k];
      s_ci[gl][sb + k] = ti[k];
    }
  }
  __syncthreads();

  // Block-level selection: wave w extracts top-13 for g-locals {w, w+8}
  // from 208 candidates via shfl_xor argmax butterflies.
  for (int pass = 0; pass < 2; ++pass) {
    const int mg = w + pass * 8;
    float cv[7];
    int   ci[7];
#pragma unroll
    for (int i = 0; i < 7; ++i) {
      const int slot = lane + 32 * i;
      const bool ok = slot < CAND_PER_G;
      cv[i] = ok ? s_cv[mg][slot] : neg_inf();
      ci[i] = ok ? s_ci[mg][slot] : 0x7fffffff;
    }
    for (int r = 0; r < K_TOP; ++r) {
      float bvv = cv[0];
      int   bii = ci[0];
      int   bsl = 0;
#pragma unroll
      for (int i = 1; i < 7; ++i) argmax_upd(bvv, bii, bsl, cv[i], ci[i], i);
      int owner = (lane << 3) | bsl;
#pragma unroll
      for (int off = 16; off >= 1; off >>= 1) {
        const float ov2 = __shfl_xor(bvv, off, 32);
        const int   oi2 = __shfl_xor(bii, off, 32);
        const int   oo2 = __shfl_xor(owner, off, 32);
        argmax_upd(bvv, bii, owner, ov2, oi2, oo2);
      }
      // Winner lane retires its slot (constant-index unrolled clear).
      if ((owner >> 3) == lane) {
        const int ms = owner & 7;
#pragma unroll
        for (int i = 0; i < 7; ++i)
          if (ms == i) { cv[i] = neg_inf(); ci[i] = 0x7fffffff; }
      }
      if (lane == 0) {
        const size_t o =
            ((size_t)(b * G_DIM + gbase + mg) * NSPLIT + split) * K_TOP + r;
        topv[o] = bvv;
        topi[o] = bii;
      }
    }
  }
}

// ---------------------------------------------------------------------------
// Kernel B: per batch (1 wave), merge 4x13 split candidates -> final top-13
// per g, then serial scatter over g (same-wave stores are in-order, which
// reproduces the reference scan's last-writer-wins semantics).
// ---------------------------------------------------------------------------
__global__ __launch_bounds__(32) void tal_assign(
    const float* __restrict__ topv, const int* __restrict__ topi,
    const int* __restrict__ gt_labels, const float* __restrict__ gt_bboxes,
    const float* __restrict__ mask_gt, float* __restrict__ out)
{
  const int b = blockIdx.x;
  const int lane = threadIdx.x;

  float* out_lab = out;                                 // [B,A]
  float* out_box = out + (size_t)B_DIM * A_DIM;         // [B,A,4]
  float* out_sc  = out + (size_t)B_DIM * A_DIM * 5;     // [B,A,C]

  for (int g = 0; g < G_DIM; ++g) {
    const int gi = b * G_DIM + g;
    const float mk = mask_gt[gi];
    const int   lab = gt_labels[gi];
    const float q0 = gt_bboxes[gi * 4 + 0];
    const float q1 = gt_bboxes[gi * 4 + 1];
    const float q2 = gt_bboxes[gi * 4 + 2];
    const float q3 = gt_bboxes[gi * 4 + 3];

    const int s1 = lane + 32;
    float v0 = topv[(size_t)gi * NCAND_FINAL + lane];
    int   i0 = topi[(size_t)gi * NCAND_FINAL + lane];
    float v1 = (s1 < NCAND_FINAL) ? topv[(size_t)gi * NCAND_FINAL + s1] : neg_inf();
    int   i1 = (s1 < NCAND_FINAL) ? topi[(size_t)gi * NCAND_FINAL + s1] : 0x7fffffff;
    if (lane >= NCAND_FINAL) { v0 = neg_inf(); i0 = 0x7fffffff; }

    float wv = 0.0f;
    int   wi = 0;
    for (int r = 0; r < K_TOP; ++r) {
      float bvv = v0;
      int   bii = i0;
      int   bs = 0;
      if (v1 > bvv || (v1 == bvv && i1 < bii)) { bvv = v1; bii = i1; bs = 1; }
      int owner = (lane << 1) | bs;
#pragma unroll
      for (int off = 16; off >= 1; off >>= 1) {
        const float ov2 = __shfl_xor(bvv, off, 32);
        const int   oi2 = __shfl_xor(bii, off, 32);
        const int   oo2 = __shfl_xor(owner, off, 32);
        argmax_upd(bvv, bii, owner, ov2, oi2, oo2);
      }
      if ((owner >> 1) == lane) {
        if (owner & 1) { v1 = neg_inf(); i1 = 0x7fffffff; }
        else           { v0 = neg_inf(); i0 = 0x7fffffff; }
      }
      if (lane == r) { wv = bvv; wi = bii; }
    }

    if (lane < K_TOP && mk > 0.0f) {
      out_lab[(size_t)b * A_DIM + wi] = (float)lab;
      float* bp = out_box + ((size_t)b * A_DIM + wi) * 4;
      bp[0] = q0; bp[1] = q1; bp[2] = q2; bp[3] = q3;
      out_sc[((size_t)b * A_DIM + wi) * C_DIM + lab] = wv;
    }
  }
}

// ---------------------------------------------------------------------------
// Output init: labels = NUM_CLASSES (80), bboxes = 0, scores = 0.
// ---------------------------------------------------------------------------
__global__ void tal_init(float* __restrict__ out)
{
  const long nlab = (long)B_DIM * A_DIM;
  const long ntot = nlab * (1 + 4 + C_DIM);
  long i = (long)blockIdx.x * blockDim.x + threadIdx.x;
  const long stride = (long)gridDim.x * blockDim.x;
  for (; i < ntot; i += stride) out[i] = (i < nlab) ? 80.0f : 0.0f;
}

extern "C" void kernel_launch(void* const* d_in, const int* in_sizes, int n_in,
                              void* d_out, int out_size, void* d_ws, size_t ws_size,
                              hipStream_t stream) {
  const float* pred_scores = (const float*)d_in[0];
  const float* pred_bboxes = (const float*)d_in[1];
  // d_in[2] = anchor_points (unused by reference)
  const int*   gt_labels   = (const int*)d_in[3];
  const float* gt_bboxes   = (const float*)d_in[4];
  const float* mask_gt     = (const float*)d_in[5];
  float* out = (float*)d_out;

  const size_t n_top = (size_t)B_DIM * G_DIM * NSPLIT * K_TOP;   // 53,248
  float* topv = (float*)d_ws;
  int*   topi = (int*)((char*)d_ws + n_top * sizeof(float));

  hipLaunchKernelGGL(tal_init, dim3(2048), dim3(256), 0, stream, out);
  hipLaunchKernelGGL(tal_metric_topk, dim3(B_DIM * 4 * NSPLIT), dim3(256), 0, stream,
                     pred_scores, pred_bboxes, gt_labels, gt_bboxes, mask_gt,
                     topv, topi);
  hipLaunchKernelGGL(tal_assign, dim3(B_DIM), dim3(32), 0, stream,
                     topv, topi, gt_labels, gt_bboxes, mask_gt, out);
}
